// TCNNModel_35287451304097
// MI455X (gfx1250) — matrix operand order repaired and checked by reading, not compiled, not run on hardware
//
#include <hip/hip_runtime.h>
#include <hip/hip_bf16.h>

// CDNA5 wave32 WMMA types
typedef __attribute__((ext_vector_type(16))) _Float16 v16h;
typedef __attribute__((ext_vector_type(8)))  _Float16 v8h;
typedef __attribute__((ext_vector_type(8)))  float    v8f;

// lrelu with slope 0.01:  0.505*x + 0.495*|x|  -> v_mul + v_fma (|x| is a free
// source modifier); avoids fmaxf's canonicalize and any VCC use.
__device__ __forceinline__ float lrelu_fast(float x) {
  return fmaf(0.505f, x, 0.495f * fabsf(x));
}
// exact-form lrelu for the final f32 output
__device__ __forceinline__ float lrelu_exact(float x) {
  return x > 0.0f ? x : 0.01f * x;
}

__device__ __forceinline__ float triw(float xf) {
  return fabsf(xf - floorf(xf + 0.5f)) * 4.0f - 1.0f;
}

__device__ __forceinline__ float pick4(float4 s, int i) {
  float r = s.x;
  r = (i == 1) ? s.y : r;
  r = (i == 2) ? s.z : r;
  r = (i == 3) ? s.w : r;
  return r;
}

// Bilinear interp of all 4 features at one grid level. Cell features are
// contiguous fpl=4 floats -> 16B vector gathers (grids ~111MB < 192MB L2).
__device__ __forceinline__ float4 grid_bilerp(const float* __restrict__ params,
                                              int level, float u, float v) {
  const int res = 16 << level;                            // ceil(16*2^l - 1) + 1
  const float scale = (float)(res - 1);
  const int off = (1024 * ((1 << (2 * level)) - 1)) / 3;  // 1024*(4^l-1)/3 floats
  float px = u * scale + 0.5f;
  float py = v * scale + 0.5f;
  float fx = floorf(px), fy = floorf(py);
  float wx = px - fx, wy = py - fy;
  int ix0 = (int)fx; ix0 = ix0 < 0 ? 0 : (ix0 > res - 1 ? res - 1 : ix0);
  int iy0 = (int)fy; iy0 = iy0 < 0 ? 0 : (iy0 > res - 1 ? res - 1 : iy0);
  int ix1 = ix0 + 1 < res ? ix0 + 1 : res - 1;
  int iy1 = iy0 + 1 < res ? iy0 + 1 : res - 1;
  const float4* t = (const float4*)(params + off);
  float4 f00 = t[(size_t)iy0 * res + ix0];
  float4 f10 = t[(size_t)iy0 * res + ix1];
  float4 f01 = t[(size_t)iy1 * res + ix0];
  float4 f11 = t[(size_t)iy1 * res + ix1];
  float4 r;
  r.x = (f00.x * (1.f - wx) + f10.x * wx) * (1.f - wy) + (f01.x * (1.f - wx) + f11.x * wx) * wy;
  r.y = (f00.y * (1.f - wx) + f10.y * wx) * (1.f - wy) + (f01.y * (1.f - wx) + f11.y * wx) * wy;
  r.z = (f00.z * (1.f - wx) + f10.z * wx) * (1.f - wy) + (f01.z * (1.f - wx) + f11.z * wx) * wy;
  r.w = (f00.w * (1.f - wx) + f10.w * wx) * (1.f - wy) + (f01.w * (1.f - wx) + f11.w * wx) * wy;
  return r;
}

// take_along_axis: 4 consecutive columns starting at c0 = trunc((L-1-clipped)*4);
// may straddle two adjacent levels -> interpolate at most 2 levels.
__device__ __forceinline__ void grid_feats(const float* __restrict__ params, int L,
                                           float mips, float u, float v, float o[4]) {
  float clipped = fminf(mips, (float)(L - 1));
  int c0 = (int)(((float)(L - 1) - clipped) * 4.0f);
  int la = c0 >> 2;
  int lb = (c0 + 3) >> 2;
  float4 fa = grid_bilerp(params, la, u, v);
  float4 fb = fa;
  if (lb != la) fb = grid_bilerp(params, lb, u, v);
#pragma unroll
  for (int j = 0; j < 4; ++j) {
    int c = c0 + j;
    float4 s = ((c >> 2) == la) ? fa : fb;
    o[j] = pick4(s, c & 3);
  }
}

// ---- WMMA fragment builders (CDNA5 wave32 layouts, CDNA5 ISA) ----

// A 16x32 f16 from row-major act LDS: lane L holds row L&15; lane-half selects
// the 16-wide K half. Each half of the fragment is 8 contiguous halves -> ds_load_b128.
__device__ __forceinline__ v16h load_a_frag(const _Float16* lds, int rowBase,
                                            int k0, int lane) {
  const _Float16* p = lds + (rowBase + (lane & 15)) * 64 + k0 + ((lane >> 4) << 4);
  v8h lo = *(const v8h*)p;          // elems 0..7  : K = g .. g+7
  v8h hi = *(const v8h*)(p + 8);    // elems 8..15 : K = g+8 .. g+15
  return __builtin_shufflevector(lo, hi, 0, 1, 2, 3, 4, 5, 6, 7,
                                 8, 9, 10, 11, 12, 13, 14, 15);
}

// B 32x16 f16 from LDS pre-swizzled into per-lane fragment order:
// frag base + lane*16 halves, 32B contiguous -> two ds_load_b128.
__device__ __forceinline__ v16h load_b_sw(const _Float16* w, int frag, int lane) {
  const _Float16* p = w + frag * 512 + lane * 16;
  v8h lo = *(const v8h*)p;
  v8h hi = *(const v8h*)(p + 8);
  return __builtin_shufflevector(lo, hi, 0, 1, 2, 3, 4, 5, 6, 7,
                                 8, 9, 10, 11, 12, 13, 14, 15);
}

__device__ __forceinline__ v8f wmma16(v16h a, v16h b, v8f c) {
  return __builtin_amdgcn_wmma_f32_16x16x32_f16(false, a, false, b, (short)0, c,
                                                false, false);
}

// C/D 16x16 f32: VGPR v holds row M = v + 8*(lane>=16), col N = lane&15.
__device__ __forceinline__ void store_d_act(_Float16* lds, int rowBase,
                                            int n0, int lane, v8f d) {
  int nn = n0 + (lane & 15);
  int mo = (lane >> 4) << 3;
#pragma unroll
  for (int vv = 0; vv < 8; ++vv) {
    lds[(rowBase + mo + vv) * 64 + nn] = (_Float16)lrelu_fast(d[vv]);
  }
}

__global__ __launch_bounds__(256) void tcnn_fused(
    const float* __restrict__ X, const float* __restrict__ G0,
    const float* __restrict__ G1, const float* __restrict__ Win,
    const float* __restrict__ Wh, const float* __restrict__ Wout,
    float* __restrict__ out, int n) {
  // Weights stored pre-swizzled in B-fragment per-lane order:
  // idx = ((kt*NT + nt)*32 + lane)*16 + i  ->  W[kt*32 + (lane>=16)*16 + i][nt*16 + (lane&15)]
  __shared__ __align__(16) _Float16 sWin[2 * 4 * 512];  // K padded 33->64
  __shared__ __align__(16) _Float16 sWh[2 * 4 * 512];
  __shared__ __align__(16) _Float16 sWout[2 * 1 * 512]; // N padded 11->16
  __shared__ __align__(16) _Float16 sAct[256 * 64];     // row-major activations

  const int tid = threadIdx.x;

  // Stage weights (f32 -> f16), swizzled.
  for (int idx = tid; idx < 4096; idx += 256) {
    int i = idx & 15;
    int lane = (idx >> 4) & 31;
    int frag = idx >> 9;          // kt*4 + nt
    int kt = frag >> 2, nt = frag & 3;
    int k = kt * 32 + ((lane >> 4) << 4) + i;
    int nn = nt * 16 + (lane & 15);
    sWin[idx] = (k < 33) ? (_Float16)Win[k * 64 + nn] : (_Float16)0.0f;
    sWh[idx] = (_Float16)Wh[k * 64 + nn];
  }
  for (int idx = tid; idx < 1024; idx += 256) {
    int i = idx & 15;
    int lane = (idx >> 4) & 31;
    int kt = idx >> 9;
    int k = kt * 32 + ((lane >> 4) << 4) + i;
    int nn = lane & 15;
    sWout[idx] = (nn < 11) ? (_Float16)Wout[k * 11 + nn] : (_Float16)0.0f;
  }

  // Per-thread feature build in registers, then 8x ds_store_b128.
  const int s = blockIdx.x * 256 + tid;
  float xu = 0.0f, xv = 0.0f, lod = 0.0f;
  if (s < n) {
    xu = X[s * 3 + 0];
    xv = X[s * 3 + 1];
    lod = X[s * 3 + 2];
  }
  float u = xu - floorf(xu);
  float v = xv - floorf(xv);
  float mips = lod * 7.0f;  // NUM_LODS - NUM_SAMPLED_LODS

  _Float16 f[64];
#pragma unroll
  for (int q = 0; q < 12; ++q) {
    float fr = (float)(1 << q);
    f[q]      = (_Float16)triw(u * fr);
    f[12 + q] = (_Float16)triw(v * fr);
  }
  {
    float g0[4], g1[4];
    grid_feats(G0, 8, mips, u, v, g0);
    grid_feats(G1, 7, mips, u, v, g1);
#pragma unroll
    for (int j = 0; j < 4; ++j) {
      f[24 + j] = (_Float16)g0[j];
      f[28 + j] = (_Float16)g1[j];
    }
  }
  f[32] = (_Float16)lod;
#pragma unroll
  for (int k = 33; k < 64; ++k) f[k] = (_Float16)0.0f;

  {
    _Float16* row = sAct + tid * 64;
#pragma unroll
    for (int c = 0; c < 8; ++c) {
      v8h chunk;
#pragma unroll
      for (int e = 0; e < 8; ++e) chunk[e] = f[c * 8 + e];
      *(v8h*)(row + c * 8) = chunk;
    }
  }

  __syncthreads();

  const int wave = tid >> 5;
  const int lane = tid & 31;
  const int blockS0 = blockIdx.x * 256;

#pragma unroll
  for (int t = 0; t < 2; ++t) {
    const int rowBase = wave * 32 + t * 16;  // wave-private 16-row tile

    // ---- layer 1: 33(pad 64) -> 64 ----
    v16h a0 = load_a_frag(sAct, rowBase, 0, lane);
    v16h a1 = load_a_frag(sAct, rowBase, 32, lane);
    v8f acc[4];
#pragma unroll
    for (int nt = 0; nt < 4; ++nt) {
      v8f c = {};
      c = wmma16(a0, load_b_sw(sWin, 0 * 4 + nt, lane), c);
      c = wmma16(a1, load_b_sw(sWin, 1 * 4 + nt, lane), c);
      acc[nt] = c;
    }
#pragma unroll
    for (int nt = 0; nt < 4; ++nt) store_d_act(sAct, rowBase, nt * 16, lane, acc[nt]);

    // ---- layer 2: 64 -> 64 ----
    a0 = load_a_frag(sAct, rowBase, 0, lane);
    a1 = load_a_frag(sAct, rowBase, 32, lane);
#pragma unroll
    for (int nt = 0; nt < 4; ++nt) {
      v8f c = {};
      c = wmma16(a0, load_b_sw(sWh, 0 * 4 + nt, lane), c);
      c = wmma16(a1, load_b_sw(sWh, 1 * 4 + nt, lane), c);
      acc[nt] = c;
    }
#pragma unroll
    for (int nt = 0; nt < 4; ++nt) store_d_act(sAct, rowBase, nt * 16, lane, acc[nt]);

    // ---- layer 3: 64 -> 11 (pad 16) ----
    a0 = load_a_frag(sAct, rowBase, 0, lane);
    a1 = load_a_frag(sAct, rowBase, 32, lane);
    v8f c = {};
    c = wmma16(a0, load_b_sw(sWout, 0, lane), c);
    c = wmma16(a1, load_b_sw(sWout, 1, lane), c);

    const int nn = lane & 15;
    const int mo = (lane >> 4) << 3;
    if (nn < 11) {
#pragma unroll
      for (int vv = 0; vv < 8; ++vv) {
        int gRow = blockS0 + rowBase + mo + vv;
        if (gRow < n) {
          out[gRow * 11 + nn] = lrelu_exact(c[vv]);
        }
      }
    }
  }
}

extern "C" void kernel_launch(void* const* d_in, const int* in_sizes, int n_in,
                              void* d_out, int out_size, void* d_ws, size_t ws_size,
                              hipStream_t stream) {
  const float* X   = (const float*)d_in[0];
  const float* G0  = (const float*)d_in[1];
  const float* G1  = (const float*)d_in[2];
  const float* Win = (const float*)d_in[3];
  const float* Wh  = (const float*)d_in[4];
  const float* Wo  = (const float*)d_in[5];
  float* out = (float*)d_out;
  int n = in_sizes[0] / 3;
  int blocks = (n + 255) / 256;
  tcnn_fused<<<blocks, 256, 0, stream>>>(X, G0, G1, Win, Wh, Wo, out, n);
}
